// AudioFeatEncoder_12575664242785
// MI455X (gfx1250) — matrix operand order repaired and checked by reading, not compiled, not run on hardware
//
#include <hip/hip_runtime.h>
#include <hip/hip_bf16.h>

typedef __bf16 bf16;
typedef bf16  v16bf __attribute__((ext_vector_type(16)));
typedef bf16  v8bf  __attribute__((ext_vector_type(8)));
typedef float v8f   __attribute__((ext_vector_type(8)));

#define BATCH 2
#define SEQL  4096
#define MT 128
#define NT 64
#define KT 64
#define KPAD 72   // 64 + 8 elems pad -> row stride 144 B (16B aligned)

// Build a v16bf fragment from two 16B-aligned LDS runs of 8 bf16 each.
static __device__ __forceinline__ v16bf ld_frag2(const bf16* lo, const bf16* hi) {
  v8bf a = *(const v8bf*)lo;
  v8bf b = *(const v8bf*)hi;
  return __builtin_shufflevector(a, b, 0,1,2,3,4,5,6,7,8,9,10,11,12,13,14,15);
}

// ---------------------------------------------------------------------------
// Tiled bf16 WMMA GEMM: Y(M,P) = W(M,K) @ X(K,P) + bias, optional ReLU,
// output either bf16 or f32.  W fp32 in global (converted on LDS stage).
// Block = 256 threads = 8 waves; tile 128(M) x 64(N); K-step 64 (2x32 WMMA).
// X tile is stored transposed in LDS so each lane's B fragment is two
// contiguous 16-byte k runs (2x ds_load_b128).
// ---------------------------------------------------------------------------
__global__ __launch_bounds__(256) void gemm_bias_act_wmma(
    const float* __restrict__ W, const bf16* __restrict__ X,
    const float* __restrict__ bias, void* __restrict__ Y,
    int M, int K, int P, int relu, int out_bf16)
{
  __shared__ bf16 sW [MT][KPAD];   // row-major (m, k)
  __shared__ bf16 sXt[NT][KPAD];   // transposed (n, k)
  const int tid  = threadIdx.x;
  const int wave = tid >> 5;
  const int lane = tid & 31;
  const int m0 = blockIdx.y * MT;
  const int n0 = blockIdx.x * NT;
  const int half8  = (lane >> 4) << 3;   // 0 or 8
  const int half16 = (lane >> 4) << 4;   // 0 or 16
  const int lr = lane & 15;
  const int mrow = (wave << 4) + lr;

  v8f acc[4] = {};

  for (int k0 = 0; k0 < K; k0 += KT) {
    // Stage W tile (128 x 64): float4 global loads, packed 4xbf16 8B LDS store.
    for (int i = tid; i < MT * KT / 4; i += 256) {
      int m  = i >> 4;
      int k4 = (i & 15) << 2;
      const float4 f = *(const float4*)&W[(size_t)(m0 + m) * K + (k0 + k4)];
      union { bf16 h[4]; unsigned long long u; } u;
      u.h[0] = (bf16)f.x; u.h[1] = (bf16)f.y; u.h[2] = (bf16)f.z; u.h[3] = (bf16)f.w;
      *(unsigned long long*)&sW[m][k4] = u.u;
    }
    // Stage X tile (64 x 64) transposed: 8B global loads, 4 scattered b16 stores.
    for (int i = tid; i < KT * NT / 4; i += 256) {
      int k  = i >> 4;
      int n4 = (i & 15) << 2;
      union { bf16 h[4]; unsigned long long u; } u;
      u.u = *(const unsigned long long*)&X[(size_t)(k0 + k) * P + (n0 + n4)];
      sXt[n4 + 0][k] = u.h[0];
      sXt[n4 + 1][k] = u.h[1];
      sXt[n4 + 2][k] = u.h[2];
      sXt[n4 + 3][k] = u.h[3];
    }
    if (k0 + KT < K) {
      __builtin_prefetch(&X[(size_t)(k0 + KT) * P + n0], 0, 1);
      __builtin_prefetch(&W[(size_t)m0 * K + (k0 + KT)], 0, 1);
    }
    __syncthreads();

    #pragma unroll
    for (int ks = 0; ks < KT; ks += 32) {
      // A fragment: elements 0..7 = K half8..+7, 8..15 = K 16+half8..+7
      v16bf afrag = ld_frag2(&sW[mrow][ks + half8], &sW[mrow][ks + 16 + half8]);
      #pragma unroll
      for (int nt = 0; nt < 4; ++nt) {
        int ncol = (nt << 4) + lr;
        // B fragment: elements 0..7 = K half8..+7, 8..15 = K 16+half8..+7
        v16bf bfrag = ld_frag2(&sXt[ncol][ks + half8], &sXt[ncol][ks + 16 + half8]);
        acc[nt] = __builtin_amdgcn_wmma_f32_16x16x32_bf16(
            false, afrag, false, bfrag, (short)0, acc[nt], false, false);
      }
    }
    __syncthreads();
  }

  const int mbase = m0 + (wave << 4) + half8;
  #pragma unroll
  for (int nt = 0; nt < 4; ++nt) {
    int n = n0 + (nt << 4) + lr;
    #pragma unroll
    for (int r = 0; r < 8; ++r) {
      int m = mbase + r;
      float vv = acc[nt][r] + bias[m];
      if (relu) vv = fmaxf(vv, 0.f);
      if (out_bf16) ((bf16*)Y)[(size_t)m * P + n] = (bf16)vv;
      else          ((float*)Y)[(size_t)m * P + n] = vv;
    }
  }
}

// ---------------------------------------------------------------------------
// Frontend: x[c, p] = audio[p] * w[c] + b[c]   (p over B*F*L), bf16 out
// ---------------------------------------------------------------------------
__global__ void frontend_kernel(const float* __restrict__ audio,
                                const float* __restrict__ w,
                                const float* __restrict__ b,
                                bf16* __restrict__ y, long long npos)
{
  long long i = (long long)blockIdx.x * 256 + threadIdx.x;
  if (i >= npos * 128) return;
  int c = (int)(i / npos);
  long long p = i % npos;
  y[i] = (bf16)(audio[p] * w[c] + b[c]);
}

// Depthwise 5-tap freq conv + bias + relu; layout (C,B,F,L), zero pad 2
__global__ void dwconv_freq_relu_kernel(const bf16* __restrict__ x,
    const float* __restrict__ wd, const float* __restrict__ bd,
    bf16* __restrict__ y, int C, int Bn, int F, int L, long long total)
{
  long long i = (long long)blockIdx.x * 256 + threadIdx.x;
  if (i >= total) return;
  int l = (int)(i % L);
  long long r = i / L;
  int f = (int)(r % F); r /= F;
  int b = (int)(r % Bn);
  int c = (int)(r / Bn);
  float acc = bd[c];
  #pragma unroll
  for (int j = 0; j < 5; ++j) {
    int fi = f + j - 2;
    if (fi >= 0 && fi < F)
      acc += (float)x[(((size_t)c * Bn + b) * F + fi) * (size_t)L + l] * wd[c * 5 + j];
  }
  y[i] = (bf16)fmaxf(acc, 0.f);
}

// Max-pool over freq, window=stride=s; in (C,B,Fi,L) -> out (C,B,Fo,L)
__global__ void maxpool_freq_kernel(const bf16* __restrict__ x, bf16* __restrict__ y,
                                    int C, int Bn, int Fo, int s, int L, long long total)
{
  long long i = (long long)blockIdx.x * 256 + threadIdx.x;
  if (i >= total) return;
  int l = (int)(i % L);
  long long r = i / L;
  int fo = (int)(r % Fo); r /= Fo;
  int b = (int)(r % Bn);
  int c = (int)(r / Bn);
  int Fi = Fo * s;
  float m = -3.0e38f;
  for (int k = 0; k < s; ++k)
    m = fmaxf(m, (float)x[(((size_t)c * Bn + b) * Fi + fo * s + k) * (size_t)L + l]);
  y[i] = (bf16)m;
}

__global__ void bf16_to_f32_kernel(const bf16* __restrict__ x, float* __restrict__ y, long long n)
{
  long long i = (long long)blockIdx.x * 256 + threadIdx.x;
  if (i < n) y[i] = (float)x[i];
}

// Instance norm over L per (c,b) row; fp32 in, bf16 out
__global__ __launch_bounds__(256) void inorm_kernel(const float* __restrict__ x,
                                                    bf16* __restrict__ y, int L)
{
  const float* row = x + (size_t)blockIdx.x * L;
  bf16* orow = y + (size_t)blockIdx.x * L;
  float s = 0.f, ss = 0.f;
  for (int t = threadIdx.x; t < L; t += 256) { float v = row[t]; s += v; ss += v * v; }
  __shared__ float rs[256], rss[256];
  rs[threadIdx.x] = s; rss[threadIdx.x] = ss;
  __syncthreads();
  for (int off = 128; off > 0; off >>= 1) {
    if (threadIdx.x < off) { rs[threadIdx.x] += rs[threadIdx.x + off];
                             rss[threadIdx.x] += rss[threadIdx.x + off]; }
    __syncthreads();
  }
  float mean = rs[0] / L;
  float var  = rss[0] / L - mean * mean;
  float inv  = rsqrtf(var + 1e-5f);
  for (int t = threadIdx.x; t < L; t += 256)
    orow[t] = (bf16)((row[t] - mean) * inv);
}

// Depthwise 3-tap time conv + bias + relu; layout (C,B,L), zero pad 1
__global__ void dwconv1d_relu_kernel(const bf16* __restrict__ x,
    const float* __restrict__ wd, const float* __restrict__ bd,
    bf16* __restrict__ y, int Bn, int L, long long total)
{
  long long i = (long long)blockIdx.x * 256 + threadIdx.x;
  if (i >= total) return;
  int l = (int)(i % L);
  long long cb = i / L;
  int c = (int)(cb / Bn);
  float acc = bd[c];
  #pragma unroll
  for (int j = 0; j < 3; ++j) {
    int t = l + j - 1;
    if (t >= 0 && t < L)
      acc += (float)x[cb * (size_t)L + t] * wd[c * 3 + j];
  }
  y[i] = (bf16)fmaxf(acc, 0.f);
}

// minGRU bidirectional scan: y (4096,B,L) fp32 -> h (2048,B,L) bf16.
__global__ __launch_bounds__(256) void mingru_scan_kernel(
    const float* __restrict__ y, bf16* __restrict__ h, int Bn, int L)
{
  int slot = blockIdx.x;
  int b = blockIdx.y;
  int rev = (slot >= 1024);
  int hch = rev ? (slot + 1024) : slot;
  int zch = hch + 1024;
  const float* yh = y + ((size_t)hch * Bn + b) * L;
  const float* yz = y + ((size_t)zch * Bn + b) * L;
  bf16* out = h + ((size_t)slot * Bn + b) * L;

  const int t0 = threadIdx.x * 16;
  float la[16], lb[16];
  float A = 1.f, Bv = 0.f;
  #pragma unroll
  for (int j = 0; j < 16; ++j) {
    int t = t0 + j;
    int ts = rev ? (L - 1 - t) : t;
    float ht = yh[ts];
    float z  = 1.f / (1.f + __expf(-yz[ts]));
    float a  = 1.f - z;
    float bb = z * ht;
    la[j] = a; lb[j] = bb;
    A  = a * A;
    Bv = a * Bv + bb;
  }
  __shared__ float sA[256], sB[256];
  sA[threadIdx.x] = A; sB[threadIdx.x] = Bv;
  __syncthreads();
  for (int off = 1; off < 256; off <<= 1) {
    float pA = 1.f, pB = 0.f;
    if (threadIdx.x >= off) { pA = sA[threadIdx.x - off]; pB = sB[threadIdx.x - off]; }
    __syncthreads();
    float cA = sA[threadIdx.x], cB = sB[threadIdx.x];
    sA[threadIdx.x] = pA * cA;
    sB[threadIdx.x] = cA * pB + cB;
    __syncthreads();
  }
  float hprev = (threadIdx.x > 0) ? sB[threadIdx.x - 1] : 0.f;
  #pragma unroll
  for (int j = 0; j < 16; ++j) {
    hprev = la[j] * hprev + lb[j];
    int t = t0 + j;
    int ts = rev ? (L - 1 - t) : t;
    out[ts] = (bf16)hprev;
  }
}

__global__ void resadd_kernel(float* __restrict__ x, const float* __restrict__ d, long long n)
{
  long long i = (long long)blockIdx.x * 256 + threadIdx.x;
  if (i < n) x[i] += d[i];
}

// (C,B,L) -> (B,C,L) fp32 final output
__global__ void out_transpose_kernel(const float* __restrict__ x, float* __restrict__ out,
                                     int Bn, int C, int L)
{
  long long i = (long long)blockIdx.x * 256 + threadIdx.x;
  long long total = (long long)Bn * C * L;
  if (i >= total) return;
  int l = (int)(i % L);
  long long r = i / L;
  int c = (int)(r % C);
  int b = (int)(r / C);
  out[i] = x[((size_t)c * Bn + b) * (size_t)L + l];
}

static inline unsigned cdiv256(long long n) { return (unsigned)((n + 255) / 256); }

extern "C" void kernel_launch(void* const* d_in, const int* in_sizes, int n_in,
                              void* d_out, int out_size, void* d_ws, size_t ws_size,
                              hipStream_t stream)
{
  (void)in_sizes; (void)n_in; (void)out_size; (void)ws_size;

  const float* audio = (const float*)d_in[0];
  const float* fe_w  = (const float*)d_in[1];
  const float* fe_b  = (const float*)d_in[2];

  // ---- workspace carve ----
  const size_t BIGE = 256ull * BATCH * 80 * SEQL;   // largest bf16 stage
  char* p = (char*)d_ws;
  bf16*  bufA   = (bf16*)p;  p += BIGE * sizeof(bf16);
  bf16*  bufB   = (bf16*)p;  p += BIGE * sizeof(bf16);
  float* xf     = (float*)p; p += 1024ull * BATCH * SEQL * sizeof(float);
  bf16*  nrm    = (bf16*)p;  p += 1024ull * BATCH * SEQL * sizeof(bf16);
  bf16*  dwb    = (bf16*)p;  p += 1024ull * BATCH * SEQL * sizeof(bf16);
  float* y4096  = (float*)p; p += 4096ull * BATCH * SEQL * sizeof(float);
  bf16*  h2048  = (bf16*)p;  p += 2048ull * BATCH * SEQL * sizeof(bf16);
  float* ydelta = (float*)p; p += 1024ull * BATCH * SEQL * sizeof(float);

  // ---- frontend: 1 -> 128 channels ----
  {
    long long npos = (long long)BATCH * 80 * SEQL;
    frontend_kernel<<<cdiv256(npos * 128), 256, 0, stream>>>(audio, fe_w, fe_b, bufA, npos);
  }

  // ---- frequency blocks ----
  bf16* cur = bufA; bf16* alt = bufB;
  int dc = 128, F = 80;
  const int SCL[3] = {4, 4, 5};
  for (int kblk = 0; kblk < 3; ++kblk) {
    int dh = dc * 2, s = SCL[kblk];
    int base = 3 + 8 * kblk;
    const float* w1 = (const float*)d_in[base + 0];
    const float* b1 = (const float*)d_in[base + 1];
    const float* wd = (const float*)d_in[base + 2];
    const float* bd = (const float*)d_in[base + 3];
    const float* w2 = (const float*)d_in[base + 4];
    const float* b2 = (const float*)d_in[base + 5];
    const float* w3 = (const float*)d_in[base + 6];
    const float* b3 = (const float*)d_in[base + 7];

    long long P = (long long)BATCH * F * SEQL;
    // h1 = W1 @ x + b1   (dh, P)  -> alt  (no act)
    gemm_bias_act_wmma<<<dim3((unsigned)(P / NT), (unsigned)(dh / MT)), 256, 0, stream>>>(
        w1, cur, b1, alt, dh, dc, (int)P, 0, 1);
    // h2 = relu(dwconv_freq(h1)) -> cur
    dwconv_freq_relu_kernel<<<cdiv256((long long)dh * P), 256, 0, stream>>>(
        alt, wd, bd, cur, dh, BATCH, F, SEQL, (long long)dh * P);
    // h3 = relu(W2 @ h2 + b2)   (dc, P) -> alt
    gemm_bias_act_wmma<<<dim3((unsigned)(P / NT), (unsigned)(dc / MT)), 256, 0, stream>>>(
        w2, cur, b2, alt, dc, dh, (int)P, 1, 1);
    // h4 = maxpool_freq(h3, s)  (dc, B, F/s, L) -> cur
    int Fo = F / s;
    long long tot4 = (long long)dc * BATCH * Fo * SEQL;
    maxpool_freq_kernel<<<cdiv256(tot4), 256, 0, stream>>>(
        alt, cur, dc, BATCH, Fo, s, SEQL, tot4);
    F = Fo;
    long long P2 = (long long)BATCH * F * SEQL;
    // x = W3 @ h4 + b3   (dh, P2) -> alt
    gemm_bias_act_wmma<<<dim3((unsigned)(P2 / NT), (unsigned)(dh / MT)), 256, 0, stream>>>(
        w3, cur, b3, alt, dh, dc, (int)P2, 0, 1);
    bf16* t = cur; cur = alt; alt = t;
    dc = dh;
  }

  // ---- to fp32 residual stream: x (1024, B*L) ----
  {
    long long n = 1024LL * BATCH * SEQL;
    bf16_to_f32_kernel<<<cdiv256(n), 256, 0, stream>>>(cur, xf, n);
  }

  // ---- time blocks ----
  for (int t = 0; t < 2; ++t) {
    int base = 27 + 6 * t;
    const float* wd = (const float*)d_in[base + 0];
    const float* bd = (const float*)d_in[base + 1];
    const float* w1 = (const float*)d_in[base + 2];
    const float* b1 = (const float*)d_in[base + 3];
    const float* w2 = (const float*)d_in[base + 4];
    const float* b2 = (const float*)d_in[base + 5];

    inorm_kernel<<<1024 * BATCH, 256, 0, stream>>>(xf, nrm, SEQL);
    long long n1 = 1024LL * BATCH * SEQL;
    dwconv1d_relu_kernel<<<cdiv256(n1), 256, 0, stream>>>(nrm, wd, bd, dwb, BATCH, SEQL, n1);

    long long P = (long long)BATCH * SEQL;   // 8192
    // y = W1 @ dwb + b1  (4096, P) fp32
    gemm_bias_act_wmma<<<dim3((unsigned)(P / NT), 4096 / MT), 256, 0, stream>>>(
        w1, dwb, b1, y4096, 4096, 1024, (int)P, 0, 0);
    // bidirectional minGRU scan -> h (2048, P) bf16
    mingru_scan_kernel<<<dim3(2048, BATCH), 256, 0, stream>>>(y4096, h2048, BATCH, SEQL);
    // delta = W2 @ h + b2  (1024, P) fp32
    gemm_bias_act_wmma<<<dim3((unsigned)(P / NT), 1024 / MT), 256, 0, stream>>>(
        w2, h2048, b2, ydelta, 1024, 2048, (int)P, 0, 0);
    resadd_kernel<<<cdiv256(1024LL * P), 256, 0, stream>>>(xf, ydelta, 1024LL * P);
  }

  // ---- final (C,B,L) -> (B,C,L) ----
  out_transpose_kernel<<<cdiv256(1024LL * BATCH * SEQL), 256, 0, stream>>>(
      xf, (float*)d_out, BATCH, 1024, SEQL);
}
